// Transformer_Block_26259430048682
// MI455X (gfx1250) — compile-verified, matrix operand order checked by reading, not run on hardware
//
#include <hip/hip_runtime.h>
#include <math.h>

// ---------------- problem constants ----------------
constexpr int B   = 4;
constexpr int C   = 128;
constexpr int H   = 128;
constexpr int W   = 128;
constexpr int HW  = H * W;          // 16384
constexpr int NPIX= B * HW;         // 65536
constexpr int NH  = 8;
constexpr int D   = 16;             // C / NH
constexpr int C3  = 3 * C;          // 384
constexpr int G   = 2;
constexpr int CG  = G * C;          // 256
constexpr int CG2 = 2 * G * C;      // 512

// ---------------- types / helpers ----------------
typedef __bf16 bf16_t;
typedef __attribute__((ext_vector_type(16))) __bf16 v16bf;
typedef __attribute__((ext_vector_type(8)))  float  v8f;

__device__ __attribute__((always_inline)) inline bf16_t f2bf(float f) {
    unsigned u = __builtin_bit_cast(unsigned, f);
    unsigned r = u + 0x7FFFu + ((u >> 16) & 1u);           // round-to-nearest-even
    unsigned short h = (unsigned short)(r >> 16);
    return __builtin_bit_cast(bf16_t, h);
}
__device__ __attribute__((always_inline)) inline float bf2f(bf16_t b) {
    unsigned short h = __builtin_bit_cast(unsigned short, b);
    unsigned u = ((unsigned)h) << 16;
    return __builtin_bit_cast(float, u);
}
__device__ __attribute__((always_inline)) inline v16bf load_v16bf(const bf16_t* p) {
    v16bf v;
    __builtin_memcpy(&v, p, sizeof(v));
    return v;
}
__device__ __attribute__((always_inline)) inline v8f wmma_bf16(v16bf a, v16bf b, v8f c) {
    // v_wmma_f32_16x16x32_bf16: D = A(16x32) * B(32x16) + C, fp32 accumulate
    return __builtin_amdgcn_wmma_f32_16x16x32_bf16(false, a, false, b, (short)0, c,
                                                   false, false);
}

// ---------------- 0: zero stats (sumsq + gram accumulators) ----------------
__global__ void zero_kernel(float* __restrict__ p, int n) {
    int i = blockIdx.x * 256 + threadIdx.x;
    if (i < n) p[i] = 0.0f;
}

// ---------------- 1: channel LayerNorm (NCHW, over C) -> bf16 ----------------
// block: 256 threads = 8 channel-slices x 32 pixels
__global__ __launch_bounds__(256)
void ln_kernel(const float* __restrict__ x, const float* __restrict__ gamma,
               const float* __restrict__ beta, bf16_t* __restrict__ y) {
    __shared__ float tile[C][32];
    __shared__ float redS[8][32];
    __shared__ float redQ[8][32];
    __shared__ float muS[32], rsS[32];

    const int tid = threadIdx.x;
    const int px  = tid & 31;
    const int sl  = tid >> 5;
    const int pg  = blockIdx.x * 32 + px;       // pixel in [0, B*HW)
    const int b   = pg >> 14;                   // /16384
    const int p   = pg & (HW - 1);

    float s = 0.0f, q = 0.0f;
    #pragma unroll
    for (int i = 0; i < 16; ++i) {
        int c = sl * 16 + i;
        float v = x[((size_t)(b * C + c)) * HW + p];
        tile[c][px] = v;
        s += v; q += v * v;
    }
    redS[sl][px] = s; redQ[sl][px] = q;
    __syncthreads();
    if (sl == 0) {
        float ts = 0.0f, tq = 0.0f;
        #pragma unroll
        for (int i = 0; i < 8; ++i) { ts += redS[i][px]; tq += redQ[i][px]; }
        float mu  = ts * (1.0f / C);
        float var = tq * (1.0f / C) - mu * mu;
        muS[px] = mu;
        rsS[px] = rsqrtf(var + 1e-5f);
    }
    __syncthreads();
    const float mu = muS[px], rs = rsS[px];
    #pragma unroll
    for (int i = 0; i < 16; ++i) {
        int c = sl * 16 + i;
        float v = (tile[c][px] - mu) * rs * gamma[c] + beta[c];
        y[((size_t)(b * C + c)) * HW + p] = f2bf(v);
    }
}

// ---------------- 2: bf16 WMMA GEMM for 1x1 convs ----------------
// out[b, m, pix] = sum_k A[m,k] * act[b, k, pix]  (+ residual)
// block tile: BM=64 x BN=128, K-step 32; 8 waves (4 in M x 2 in N)
constexpr int BM = 64, BN = 128, BK = 32;

template<int M_TOT, int K_TOT, bool A_PER_BATCH, bool RES, bool OUT_BF16>
__global__ __launch_bounds__(256)
void gemm_bf16_kernel(const float* __restrict__ Aw,
                      const bf16_t* __restrict__ Bact, int bStride, int bOff,
                      const float* __restrict__ Res,
                      float* __restrict__ OutF, bf16_t* __restrict__ OutB) {
    __shared__ bf16_t As[BM][BK];          // [m][k]
    __shared__ bf16_t BsT[BN][BK];         // [n][k]  (transposed store)

    const int b   = blockIdx.z;
    const int p0  = blockIdx.x * BN;
    const int m0  = blockIdx.y * BM;
    const int tid = threadIdx.x;
    const int lane = tid & 31;
    const int wid  = tid >> 5;
    const int wm   = wid & 3;              // M sub-tile (16 rows each)
    const int wn   = wid >> 2;             // N half (64 cols each)

    const float* A = Aw + (A_PER_BATCH ? (size_t)b * M_TOT * K_TOT : 0);

    const v8f zero8 = {0.f, 0.f, 0.f, 0.f, 0.f, 0.f, 0.f, 0.f};
    v8f acc[4] = {zero8, zero8, zero8, zero8};

    // staging indices (constant across K loop)
    const int aIdx = tid * 8;
    const int am   = aIdx >> 5;            // 0..63
    const int ac   = aIdx & 31;            // 0,8,16,24
    const int br   = tid >> 3;             // 0..31 (k row)
    const int bc0  = (tid & 7) * 16;       // 0..112 (pixel chunk)

    for (int kk = 0; kk < K_TOT; kk += BK) {
        // --- stage A (weights fp32 -> bf16), 64x32 ---
        {
            const float* src = A + (size_t)(m0 + am) * K_TOT + kk + ac;
            float4 f0 = *(const float4*)(src);
            float4 f1 = *(const float4*)(src + 4);
            bf16_t* dst = &As[am][ac];
            dst[0] = f2bf(f0.x); dst[1] = f2bf(f0.y);
            dst[2] = f2bf(f0.z); dst[3] = f2bf(f0.w);
            dst[4] = f2bf(f1.x); dst[5] = f2bf(f1.y);
            dst[6] = f2bf(f1.z); dst[7] = f2bf(f1.w);
            if (kk + BK < K_TOT) __builtin_prefetch(src + BK, 0, 0);
        }
        // --- stage B (activations bf16, NCHW rows are contiguous pixels),
        //     stored transposed so WMMA fragments are contiguous 16-bf16 reads ---
        {
            const bf16_t* src =
                Bact + ((size_t)(b * bStride + bOff + kk + br)) * HW + p0 + bc0;
            bf16_t tmp[16];
            __builtin_memcpy(tmp, src, 32);
            #pragma unroll
            for (int j = 0; j < 16; ++j) BsT[bc0 + j][br] = tmp[j];
            if (kk + BK < K_TOT) __builtin_prefetch(src + (size_t)BK * HW, 0, 0);
        }
        __syncthreads();

        const int khalf = (lane >> 4) * 16;
        const int rm    = lane & 15;
        v16bf afrag = load_v16bf(&As[wm * 16 + rm][khalf]);
        #pragma unroll
        for (int ns = 0; ns < 4; ++ns) {
            v16bf bfrag = load_v16bf(&BsT[wn * 64 + ns * 16 + rm][khalf]);
            acc[ns] = wmma_bf16(afrag, bfrag, acc[ns]);
        }
        __syncthreads();
    }

    // --- epilogue: C/D layout -> lanes 0-15: M=r, lanes 16-31: M=r+8 ---
    const int colb = lane & 15;
    const int rsel = (lane >> 4) * 8;
    #pragma unroll
    for (int ns = 0; ns < 4; ++ns) {
        const int col = p0 + wn * 64 + ns * 16 + colb;
        #pragma unroll
        for (int r = 0; r < 8; ++r) {
            const int ch = m0 + wm * 16 + r + rsel;
            const size_t addr = ((size_t)(b * M_TOT + ch)) * HW + col;
            float v = acc[ns][r];
            if (RES) v += Res[addr];
            if (OUT_BF16) OutB[addr] = f2bf(v);
            else          OutF[addr] = v;
        }
    }
}

// ---------------- 3: depthwise 3x3 (+ q/k sum-of-squares) ----------------
// block: 256 threads = 2 rows x 128 cols of one (b, c) plane
template<int CH, bool SUMSQ>
__global__ __launch_bounds__(256)
void dwconv_kernel(const bf16_t* __restrict__ in, const float* __restrict__ wk,
                   bf16_t* __restrict__ out, float* __restrict__ sumsq) {
    __shared__ float red[256];
    int blk = blockIdx.x;
    const int hp = blk & 63; blk >>= 6;           // H/2 pairs
    const int c  = blk % CH;
    const int b  = blk / CH;
    const int tid = threadIdx.x;
    const int h = hp * 2 + (tid >> 7);
    const int w = tid & 127;

    const bf16_t* src = in + ((size_t)(b * CH + c)) * HW;
    float wr[9];
    #pragma unroll
    for (int i = 0; i < 9; ++i) wr[i] = wk[c * 9 + i];

    float acc = 0.0f;
    #pragma unroll
    for (int di = -1; di <= 1; ++di) {
        const int hh = h + di;
        if ((unsigned)hh < (unsigned)H) {
            #pragma unroll
            for (int dj = -1; dj <= 1; ++dj) {
                const int ww = w + dj;
                if ((unsigned)ww < (unsigned)W)
                    acc += bf2f(src[hh * W + ww]) * wr[(di + 1) * 3 + (dj + 1)];
            }
        }
    }
    out[((size_t)(b * CH + c)) * HW + h * W + w] = f2bf(acc);

    if (SUMSQ) {   // fp32 sum of squares for q (c<128) and k (128<=c<256)
        red[tid] = (c < 2 * C) ? acc * acc : 0.0f;
        __syncthreads();
        for (int st = 128; st > 0; st >>= 1) {
            if (tid < st) red[tid] += red[tid + st];
            __syncthreads();
        }
        if (tid == 0 && c < 2 * C) atomicAdd(&sumsq[b * 2 * C + c], red[0]);
    }
}

// ---------------- 4: attention Gram q*k^T via WMMA, n-split + atomics ----------
__global__ __launch_bounds__(32)
void gram_kernel(const bf16_t* __restrict__ qkv, float* __restrict__ gram) {
    int blk = blockIdx.x;
    const int sp = blk & 63; blk >>= 6;            // 64 splits of n
    const int hh = blk & 7;
    const int b  = blk >> 3;
    const int lane  = threadIdx.x;
    const int rm    = lane & 15;
    const int khalf = (lane >> 4) * 16;

    const bf16_t* qp = qkv + ((size_t)(b * C3 +     hh * D + rm)) * HW;
    const bf16_t* kp = qkv + ((size_t)(b * C3 + C + hh * D + rm)) * HW;

    v8f acc = {0.f, 0.f, 0.f, 0.f, 0.f, 0.f, 0.f, 0.f};
    const int n0 = sp * 256 + khalf;
    #pragma unroll
    for (int it = 0; it < 8; ++it) {
        v16bf a  = load_v16bf(qp + n0 + it * 32);   // A[d][n]
        v16bf bb = load_v16bf(kp + n0 + it * 32);   // B[n][e] = k[e][n]
        acc = wmma_bf16(a, bb, acc);
    }
    float* g = gram + ((size_t)(b * NH + hh)) * 256;
    #pragma unroll
    for (int r = 0; r < 8; ++r) {
        const int row = r + ((lane >> 4) << 3);
        atomicAdd(&g[row * 16 + (lane & 15)], acc[r]);
    }
}

// ---------------- 5: l2norm scales + softmax + fold proj into M[b] -----------
__global__ __launch_bounds__(256)
void softmax_proj_kernel(const float* __restrict__ gram, const float* __restrict__ sumsq,
                         const float* __restrict__ temp, const float* __restrict__ projw,
                         float* __restrict__ M) {
    __shared__ float inv[1024];                 // 1/max(||row||, 1e-12) for q,k
    __shared__ float attn[B * NH * 16 * 16];    // 32 KB
    const int tid = threadIdx.x;

    for (int i = tid; i < 1024; i += 256)
        inv[i] = 1.0f / fmaxf(sqrtf(sumsq[i]), 1e-12f);
    __syncthreads();

    for (int rid = tid; rid < 512; rid += 256) {
        const int b  = rid >> 7;
        const int hh = (rid >> 4) & 7;
        const int d  = rid & 15;
        const float iq = inv[b * 256 + hh * 16 + d];
        const float t  = temp[hh];
        float row[16];
        float mx = -1e30f;
        #pragma unroll
        for (int e = 0; e < 16; ++e) {
            float v = gram[((b * 8 + hh) * 16 + d) * 16 + e] * iq *
                      inv[b * 256 + 128 + hh * 16 + e] * t;
            row[e] = v; mx = fmaxf(mx, v);
        }
        float s = 0.0f;
        #pragma unroll
        for (int e = 0; e < 16; ++e) { row[e] = __expf(row[e] - mx); s += row[e]; }
        const float is = 1.0f / s;
        #pragma unroll
        for (int e = 0; e < 16; ++e)
            attn[((b * 8 + hh) * 16 + d) * 16 + e] = row[e] * is;
    }
    __syncthreads();

    // M[b][o][h*16+e] = sum_d proj[o][h*16+d] * attn[b,h,d,e]
    for (int idx = tid; idx < B * C * C; idx += 256) {
        const int b = idx >> 14;
        const int o = (idx >> 7) & 127;
        const int j = idx & 127;
        const int hh = j >> 4, e = j & 15;
        float s = 0.0f;
        #pragma unroll
        for (int d = 0; d < 16; ++d)
            s += projw[o * 128 + hh * 16 + d] * attn[((b * 8 + hh) * 16 + d) * 16 + e];
        M[idx] = s;
    }
}

// ---------------- 6: depthwise 3x3 pair + exact GELU gate --------------------
__global__ __launch_bounds__(256)
void dwconv_gelu_kernel(const bf16_t* __restrict__ in, const float* __restrict__ wk,
                        bf16_t* __restrict__ out) {
    int blk = blockIdx.x;
    const int hp = blk & 63; blk >>= 6;
    const int c  = blk % CG;
    const int b  = blk / CG;
    const int tid = threadIdx.x;
    const int h = hp * 2 + (tid >> 7);
    const int w = tid & 127;

    const bf16_t* s1 = in + ((size_t)(b * CG2 + c))      * HW;
    const bf16_t* s2 = in + ((size_t)(b * CG2 + CG + c)) * HW;
    float w1[9], w2[9];
    #pragma unroll
    for (int i = 0; i < 9; ++i) { w1[i] = wk[c * 9 + i]; w2[i] = wk[(CG + c) * 9 + i]; }

    float u1 = 0.0f, u2 = 0.0f;
    #pragma unroll
    for (int di = -1; di <= 1; ++di) {
        const int hh = h + di;
        if ((unsigned)hh < (unsigned)H) {
            #pragma unroll
            for (int dj = -1; dj <= 1; ++dj) {
                const int ww = w + dj;
                if ((unsigned)ww < (unsigned)W) {
                    const int o = hh * W + ww, wi = (di + 1) * 3 + (dj + 1);
                    u1 += bf2f(s1[o]) * w1[wi];
                    u2 += bf2f(s2[o]) * w2[wi];
                }
            }
        }
    }
    const float gl = 0.5f * u1 * (1.0f + erff(u1 * 0.70710678118654752f));
    out[((size_t)(b * CG + c)) * HW + h * W + w] = f2bf(gl * u2);
}

// ---------------- launch ----------------
extern "C" void kernel_launch(void* const* d_in, const int* in_sizes, int n_in,
                              void* d_out, int out_size, void* d_ws, size_t ws_size,
                              hipStream_t stream) {
    (void)in_sizes; (void)n_in; (void)out_size; (void)ws_size;
    const float* x     = (const float*)d_in[0];
    const float* ln1w  = (const float*)d_in[1];
    const float* ln1b  = (const float*)d_in[2];
    const float* temp  = (const float*)d_in[3];
    const float* qkvpw = (const float*)d_in[4];
    const float* qkvdw = (const float*)d_in[5];
    const float* projw = (const float*)d_in[6];
    const float* ln2w  = (const float*)d_in[7];
    const float* ln2b  = (const float*)d_in[8];
    const float* g1w   = (const float*)d_in[9];
    const float* gdw   = (const float*)d_in[10];
    const float* g2w   = (const float*)d_in[11];
    float* out = (float*)d_out;

    char* ws = (char*)d_ws;
    size_t off = 0;
    auto take = [&](size_t bytes) -> char* {
        char* p = ws + off;
        off = (off + bytes + 255) & ~(size_t)255;
        return p;
    };
    bf16_t* yb    = (bf16_t*)take((size_t)NPIX * C   * 2);  // ln1/ln2 output
    bf16_t* big1  = (bf16_t*)take((size_t)NPIX * CG2 * 2);  // qkv_pre, later hdn_pre
    bf16_t* big2  = (bf16_t*)take((size_t)NPIX * C3  * 2);  // qkv_dw, later gated g
    float*  x1    = (float*) take((size_t)NPIX * C   * 4);  // MDTA output (residual)
    float*  sumsq = (float*) take(1024 * 4);                // ||q||^2, ||k||^2 per (b,c)
    float*  gram  = (float*) take(B * NH * 256 * 4);        // unnormalized q k^T
    float*  Mbuf  = (float*) take((size_t)B * C * C * 4);   // proj @ blockdiag(attn)

    // sumsq and gram are contiguous (1024*4 is 256-aligned): zero both
    zero_kernel<<<(1024 + B * NH * 256 + 255) / 256, 256, 0, stream>>>(sumsq,
                                                               1024 + B * NH * 256);

    // --- MDTA ---
    ln_kernel<<<NPIX / 32, 256, 0, stream>>>(x, ln1w, ln1b, yb);
    gemm_bf16_kernel<C3, C, false, false, true>
        <<<dim3(HW / BN, C3 / BM, B), 256, 0, stream>>>(qkvpw, yb, C, 0,
                                                        nullptr, nullptr, big1);
    dwconv_kernel<C3, true><<<B * C3 * (H / 2), 256, 0, stream>>>(big1, qkvdw,
                                                                  big2, sumsq);
    gram_kernel<<<B * NH * 64, 32, 0, stream>>>(big2, gram);
    softmax_proj_kernel<<<1, 256, 0, stream>>>(gram, sumsq, temp, projw, Mbuf);
    gemm_bf16_kernel<C, C, true, true, false>
        <<<dim3(HW / BN, C / BM, B), 256, 0, stream>>>(Mbuf, big2, C3, 2 * C,
                                                       x, x1, nullptr);

    // --- GDFN ---
    ln_kernel<<<NPIX / 32, 256, 0, stream>>>(x1, ln2w, ln2b, yb);
    gemm_bf16_kernel<CG2, C, false, false, true>
        <<<dim3(HW / BN, CG2 / BM, B), 256, 0, stream>>>(g1w, yb, C, 0,
                                                         nullptr, nullptr, big1);
    dwconv_gelu_kernel<<<B * CG * (H / 2), 256, 0, stream>>>(big1, gdw, big2);
    gemm_bf16_kernel<C, CG, false, true, false>
        <<<dim3(HW / BN, C / BM, B), 256, 0, stream>>>(g2w, big2, CG, 0,
                                                       x1, out, nullptr);
}
